// MultiHeadAttention_2774548873660
// MI455X (gfx1250) — matrix operand order repaired and checked by reading, not compiled, not run on hardware
//
#include <hip/hip_runtime.h>

// ---------------- dims ----------------
#define BB 64
#define TT 256
#define CC 384
#define HH 6
#define DD 64
#define HD 384   // HH*DD == N_EMBD

// ---------------- vector types ----------------
typedef __attribute__((ext_vector_type(16))) __bf16          v16bf;
typedef __attribute__((ext_vector_type(8)))  __bf16          v8bf;
typedef __attribute__((ext_vector_type(8)))  float           v8f;
typedef __attribute__((ext_vector_type(8)))  unsigned short  v8us;

// ---------------- helpers ----------------
__device__ __forceinline__ unsigned short f2bf_bits(float f) {
    unsigned u = __float_as_uint(f);
    u += 0x7FFFu + ((u >> 16) & 1u);           // round-to-nearest-even
    return (unsigned short)(u >> 16);
}

__device__ __forceinline__ v8bf ld8(const unsigned short* p) {
    return *reinterpret_cast<const v8bf*>(p);  // 16B load (global_load_b128 / ds_load_b128)
}

__device__ __forceinline__ v8bf cvt8(const float* p) {
    v8us r;
#pragma unroll
    for (int i = 0; i < 8; ++i) r[i] = f2bf_bits(p[i]);
    return __builtin_bit_cast(v8bf, r);
}

__device__ __forceinline__ v16bf cat16(v8bf lo, v8bf hi) {
    return __builtin_shufflevector(lo, hi, 0,1,2,3,4,5,6,7,8,9,10,11,12,13,14,15);
}

__device__ __forceinline__ v8f wmma_bf16(v16bf a, v16bf b, v8f c) {
    return __builtin_amdgcn_wmma_f32_16x16x32_bf16(
        /*neg_a=*/false, a, /*neg_b=*/false, b,
        /*c_mod=*/(short)0, c, /*reuse_a=*/false, /*reuse_b=*/false);
}

__device__ __forceinline__ v8f zero8() {
    v8f z;
#pragma unroll
    for (int i = 0; i < 8; ++i) z[i] = 0.0f;
    return z;
}

// ============================================================
// Kernel 0: weight prep.  WqT/WkT/WvT[h][d][c] = W[h][c][d] (bf16),
//           WpT[n][c] = Wp[c][n] (bf16).  All counts == 147456.
// ============================================================
__global__ __launch_bounds__(256) void k_prep_w(
    const float* __restrict__ Wq, const float* __restrict__ Wk,
    const float* __restrict__ Wv, const float* __restrict__ Wp,
    unsigned short* __restrict__ wqT, unsigned short* __restrict__ wkT,
    unsigned short* __restrict__ wvT, unsigned short* __restrict__ wpT)
{
    const int NW = HH * CC * DD;   // 147456 (== HD*HD too)
    int idx = blockIdx.x * blockDim.x + threadIdx.x;
    int mat = idx / NW;
    int r   = idx - mat * NW;
    if (mat < 3) {
        int h  = r / (CC * DD);
        int cd = r - h * (CC * DD);
        int c  = cd / DD;
        int d  = cd - c * DD;
        const float* src = (mat == 0) ? Wq : (mat == 1) ? Wk : Wv;
        unsigned short* dst = (mat == 0) ? wqT : (mat == 1) ? wkT : wvT;
        dst[((size_t)h * DD + d) * CC + c] = f2bf_bits(src[r]);
    } else if (mat == 3) {
        int c = r / HD;
        int n = r - c * HD;
        wpT[(size_t)n * HD + c] = f2bf_bits(Wp[r]);
    }
}

// ============================================================
// Kernel 1: QKV projection.  One wave -> 16 rows x 64 cols for ONE
// of {q,k,v} of one (b,h): low register pressure (4 accumulators)
// for high occupancy; mat = waveId%3 so the 3 waves sharing a tile
// read the same x rows (L0/L2 locality).  q,k stored row-major
// [BH,T,D] bf16; v stored transposed [BH,D,T] bf16 for PV stage.
// ============================================================
__global__ __launch_bounds__(256) void k_qkv(
    const float* __restrict__ x,
    const unsigned short* __restrict__ wqT,
    const unsigned short* __restrict__ wkT,
    const unsigned short* __restrict__ wvT,
    unsigned short* __restrict__ qb,
    unsigned short* __restrict__ kb,
    unsigned short* __restrict__ vTb)
{
    int wid  = (blockIdx.x * blockDim.x + threadIdx.x) >> 5;  // 0..18431
    int lane = threadIdx.x & 31;
    int mat    = wid % 3;            // 0=q 1=k 2=v
    int tileId = wid / 3;            // 0..6143
    int it = tileId & 15;            // 16 row-tiles per (b,h)
    int bh = tileId >> 4;            // 0..383
    int b = bh / HH, h = bh - b * HH;
    int t0 = it * 16;
    int lo16 = lane & 15, hi = lane >> 4;

    const float* xrow = x + ((size_t)b * TT + t0 + lo16) * CC;
    const unsigned short* wT = (mat == 0) ? wqT : (mat == 1) ? wkT : wvT;
    const unsigned short* wmat = wT + (size_t)h * DD * CC;

    v8f acc[4];
#pragma unroll
    for (int j = 0; j < 4; ++j) acc[j] = zero8();

    for (int kk = 0; kk < CC; kk += 32) {
        // A fragment of x (f32 -> bf16 on the fly; VALU co-executes with WMMA)
        v16bf A = cat16(cvt8(xrow + kk + 16 * hi), cvt8(xrow + kk + 16 * hi + 8));
        // group the 4 B-fragment loads ahead of the 4 WMMAs
        v16bf Bf[4];
#pragma unroll
        for (int j = 0; j < 4; ++j) {
            const unsigned short* p = wmat + (size_t)(j * 16 + lo16) * CC + kk + 16 * hi;
            Bf[j] = cat16(ld8(p), ld8(p + 8));
        }
#pragma unroll
        for (int j = 0; j < 4; ++j) acc[j] = wmma_bf16(A, Bf[j], acc[j]);
    }

    if (mat < 2) {
        unsigned short* dst = (mat == 0) ? qb : kb;
        size_t base = ((size_t)bh * TT + t0) * DD;
#pragma unroll
        for (int j = 0; j < 4; ++j) {
            int d = j * 16 + lo16;
#pragma unroll
            for (int r = 0; r < 8; ++r) {
                int tl = r + 8 * hi;           // D layout: M = vgpr + 8*(lane>=16)
                dst[base + (size_t)tl * DD + d] = f2bf_bits(acc[j][r]);
            }
        }
    } else {
#pragma unroll
        for (int j = 0; j < 4; ++j) {
            int d = j * 16 + lo16;
#pragma unroll
            for (int r = 0; r < 8; ++r) {
                int tl = r + 8 * hi;
                vTb[((size_t)bh * DD + d) * TT + t0 + tl] = f2bf_bits(acc[j][r]);
            }
        }
    }
}

// ============================================================
// Kernel 2: causal attention.  Block = 2 waves; each wave owns its
// LDS slice (no block barrier needed; in-wave DS ordering via
// s_wait_dscnt).  scores f32 in LDS -> softmax -> probs bf16 in LDS
// -> PV via WMMA -> bf16 [B,T,HD] attn output.
// ============================================================
__global__ __launch_bounds__(64) void k_attn(
    const unsigned short* __restrict__ qb,
    const unsigned short* __restrict__ kb,
    const unsigned short* __restrict__ vTb,
    unsigned short* __restrict__ ob)
{
    __shared__ __align__(16) float          s_sc[2][16][TT];
    __shared__ __align__(16) unsigned short s_pr[2][16][TT];

    int bh = blockIdx.x;                 // 0..383
    int b = bh / HH, h = bh - b * HH;
    int w = threadIdx.x >> 5;            // wave in block: 0..1
    int lane = threadIdx.x & 31;
    int lo16 = lane & 15, hi = lane >> 4;

    const unsigned short* qh = qb  + (size_t)bh * TT * DD;
    const unsigned short* kh = kb  + (size_t)bh * TT * DD;
    const unsigned short* vh = vTb + (size_t)bh * DD * TT;

    for (int itl = 0; itl < 8; ++itl) {
        int it = w + 2 * itl;            // 0..15 query tile
        int t0 = it * 16;

        // ---- Q fragments (A-type, K = D split into 2x32) ----
        const unsigned short* qrow = qh + (size_t)(t0 + lo16) * DD;
        v16bf A0 = cat16(ld8(qrow +      16 * hi), ld8(qrow + 8 + 16 * hi));
        v16bf A1 = cat16(ld8(qrow + 32 + 16 * hi), ld8(qrow + 40 + 16 * hi));

        // ---- scores = q @ k^T (only causal tiles j <= it) ----
        for (int j = 0; j <= it; ++j) {
            int s0 = j * 16;
            const unsigned short* krow = kh + (size_t)(s0 + lo16) * DD + 16 * hi;
            v16bf B0 = cat16(ld8(krow),      ld8(krow + 8));
            v16bf B1 = cat16(ld8(krow + 32), ld8(krow + 40));
            v8f acc = zero8();
            acc = wmma_bf16(A0, B0, acc);
            acc = wmma_bf16(A1, B1, acc);
#pragma unroll
            for (int r = 0; r < 8; ++r) {
                int m = r + 8 * hi;
                int s = s0 + lo16;
                float v = acc[r] * 0.125f;                 // 1/sqrt(64)
                if (s > t0 + m) v = -1e30f;                // causal mask (diag tile)
                s_sc[w][m][s] = v;
            }
        }
        asm volatile("s_wait_dscnt 0" ::: "memory");

        // ---- softmax over valid prefix (length Ls) ----
        int Ls  = (it + 1) * 16;
        int nch = (Ls + 31) >> 5;
        int Lp  = nch * 32;
        if (lane < 16) {
            int m = lane;
            float mx = -1e30f;
            for (int s = 0; s < Ls; ++s) mx = fmaxf(mx, s_sc[w][m][s]);
            float sum = 0.0f;
            for (int s = 0; s < Ls; ++s) {
                float e = __expf(s_sc[w][m][s] - mx);
                sum += e;
                s_sc[w][m][s] = e;
            }
            float inv = 1.0f / sum;
            for (int s = 0; s < Ls; ++s) s_pr[w][m][s] = f2bf_bits(s_sc[w][m][s] * inv);
            for (int s = Ls; s < Lp; ++s) s_pr[w][m][s] = 0;   // pad partial K-chunk
        }
        asm volatile("s_wait_dscnt 0" ::: "memory");

        // ---- out = probs @ v  (A from LDS, B from v^T global) ----
        v8f o[4];
#pragma unroll
        for (int j = 0; j < 4; ++j) o[j] = zero8();
        for (int c = 0; c < nch; ++c) {
            const unsigned short* prow = &s_pr[w][lo16][c * 32 + 16 * hi];
            v16bf Ap = cat16(*reinterpret_cast<const v8bf*>(prow),
                             *reinterpret_cast<const v8bf*>(prow + 8));
            v16bf Bv[4];
#pragma unroll
            for (int j = 0; j < 4; ++j) {
                const unsigned short* vrow = vh + (size_t)(j * 16 + lo16) * TT + c * 32 + 16 * hi;
                Bv[j] = cat16(ld8(vrow), ld8(vrow + 8));
            }
#pragma unroll
            for (int j = 0; j < 4; ++j) o[j] = wmma_bf16(Ap, Bv[j], o[j]);
        }
        // ---- store attn output bf16 [B,T,H*D] ----
#pragma unroll
        for (int j = 0; j < 4; ++j) {
            int col = h * 64 + j * 16 + lo16;
#pragma unroll
            for (int r = 0; r < 8; ++r) {
                int t = t0 + r + 8 * hi;
                ob[((size_t)b * TT + t) * HD + col] = f2bf_bits(o[j][r]);
            }
        }
    }
}

// ============================================================
// Kernel 3: output projection  [B*T,384] @ Wp[384,384] + bp -> f32
// ============================================================
__global__ __launch_bounds__(256) void k_proj(
    const unsigned short* __restrict__ ob,
    const unsigned short* __restrict__ wpT,
    const float* __restrict__ bp,
    float* __restrict__ out)
{
    int wave = (blockIdx.x * blockDim.x + threadIdx.x) >> 5;  // 0..6143
    int lane = threadIdx.x & 31;
    int lo16 = lane & 15, hi = lane >> 4;
    int strip = wave % 6;          // 6 x 64-col strips
    int mt    = wave / 6;          // 0..1023 row tiles of 16
    int n0 = strip * 64;
    size_t m0 = (size_t)mt * 16;

    const unsigned short* arow = ob + (m0 + lo16) * HD;
    v8f acc[4];
#pragma unroll
    for (int j = 0; j < 4; ++j) acc[j] = zero8();

    for (int kk = 0; kk < HD; kk += 32) {
        v16bf A = cat16(ld8(arow + kk + 16 * hi), ld8(arow + kk + 16 * hi + 8));
        v16bf Bf[4];
#pragma unroll
        for (int j = 0; j < 4; ++j) {
            const unsigned short* brow = wpT + (size_t)(n0 + j * 16 + lo16) * HD + kk + 16 * hi;
            Bf[j] = cat16(ld8(brow), ld8(brow + 8));
        }
#pragma unroll
        for (int j = 0; j < 4; ++j) acc[j] = wmma_bf16(A, Bf[j], acc[j]);
    }
#pragma unroll
    for (int j = 0; j < 4; ++j) {
        int col = n0 + j * 16 + lo16;
        float bias = bp[col];
#pragma unroll
        for (int r = 0; r < 8; ++r) {
            size_t row = m0 + r + 8 * hi;
            out[row * HD + col] = acc[j][r] + bias;
        }
    }
}

// ============================================================
// launcher
// ============================================================
extern "C" void kernel_launch(void* const* d_in, const int* in_sizes, int n_in,
                              void* d_out, int out_size, void* d_ws, size_t ws_size,
                              hipStream_t stream) {
    const float* x  = (const float*)d_in[0];
    const float* Wq = (const float*)d_in[1];
    const float* Wk = (const float*)d_in[2];
    const float* Wv = (const float*)d_in[3];
    const float* Wp = (const float*)d_in[4];
    const float* bp = (const float*)d_in[5];
    float* out = (float*)d_out;

    // workspace layout (bf16 stored as ushort)
    unsigned short* ws = (unsigned short*)d_ws;
    const size_t NW   = (size_t)HH * CC * DD;      // 147456 (also HD*HD)
    const size_t NQKV = (size_t)BB * HH * TT * DD; // 6291456
    unsigned short* wqT = ws;           // NW
    unsigned short* wkT = wqT + NW;     // NW
    unsigned short* wvT = wkT + NW;     // NW
    unsigned short* wpT = wvT + NW;     // NW
    unsigned short* qb  = wpT + NW;     // NQKV
    unsigned short* kb  = qb + NQKV;    // NQKV
    unsigned short* vTb = kb + NQKV;    // NQKV
    unsigned short* obf = vTb + NQKV;   // B*T*HD == NQKV
    (void)ws_size; (void)in_sizes; (void)n_in; (void)out_size;

    // 0) weight prep: 4*147456 elements
    k_prep_w<<<(4 * 147456) / 256, 256, 0, stream>>>(Wq, Wk, Wv, Wp, wqT, wkT, wvT, wpT);
    // 1) QKV: 3*6144 = 18432 waves -> 2304 blocks x 8 waves
    k_qkv<<<2304, 256, 0, stream>>>(x, wqT, wkT, wvT, qb, kb, vTb);
    // 2) attention: one block (2 waves) per (b,h)
    k_attn<<<BB * HH, 64, 0, stream>>>(qb, kb, vTb, obf);
    // 3) output projection: 6144 waves -> 768 blocks x 8 waves
    k_proj<<<768, 256, 0, stream>>>(obf, wpT, bp, out);
}
